// Quadratic_22634477650106
// MI455X (gfx1250) — compile-verified, matrix operand order checked by reading, not run on hardware
//
#include <hip/hip_runtime.h>
#include <hip/hip_bf16.h>

typedef __attribute__((ext_vector_type(16))) _Float16 v16h;
typedef __attribute__((ext_vector_type(8)))  _Float16 v8h;
typedef __attribute__((ext_vector_type(8)))  float    v8f;
typedef __attribute__((ext_vector_type(4)))  float    v4f;

#define NX 512
#define NY 512
#define NBATCH 1024
// LDS Q-tile row pitch in f16 elements: 512 + 8 pad -> row stride 1040 B
// (multiple of 16 B for b128 loads; 260 dwords -> bank stride 4, rows k<16
// land on distinct bank-quads)
#define LDS_PITCH 520

// Stage Q[i, :, k0:k0+16] (f32, k-contiguous) into buf[k][j] as f16.
// Thread t loads a float4 at k-quad (t&3), j = (t>>2) + 64r : fully
// coalesced 64B global reads, ds_store_b16 into the transposed tile.
__device__ __forceinline__ void stage_slice(const float* __restrict__ Q,
                                            int i, int k0, int sk, int sj,
                                            _Float16* __restrict__ buf)
{
    const float* qbase = Q + (size_t)i * (NX * NY) + k0;
    #pragma unroll
    for (int r = 0; r < 8; ++r) {
        const int j = sj + 64 * r;
        v4f qv = *(const v4f*)(qbase + (size_t)j * NY + 4 * sk);
        #pragma unroll
        for (int e = 0; e < 4; ++e)
            buf[(4 * sk + e) * LDS_PITCH + j] = (_Float16)qv[e];
    }
}

// One block: 256 threads = 8 waves. Wave w owns batch rows
// [blockIdx.x*128 + 16w, +16). Block computes out[128 batches, 16 k-cols],
// reducing over the full (i,j) = 512x512 space, fused with lin + bias:
//   out[b,k] = bias[k] + sum_i x[b,i] * ( W[i,k] + sum_j Q[i,j,k] x[b,j] )
__global__ __launch_bounds__(256) void quad_fused_kernel(
    const float* __restrict__ x, const float* __restrict__ W,
    const float* __restrict__ Q, const float* __restrict__ bias,
    float* __restrict__ out)
{
    __shared__ _Float16 ldsQ[2][16 * LDS_PITCH];   // double-buffered Q tile

    const int lane = threadIdx.x & 31;
    const int wave = threadIdx.x >> 5;          // 0..7
    const int m0   = blockIdx.x * 128 + wave * 16;
    const int k0   = blockIdx.y * 16;
    const int half = lane >> 4;                 // 0 or 1 (lane group)
    const int off  = half * 8;                  // K-offset inside a 32-chunk
    const int b_row = m0 + (lane & 15);         // this lane's batch column
    const int krow  = lane & 15;                // this lane's A-matrix row (k)

    // ---- persistent B operand: x[b_row, 0:512] as f16 in WMMA B layout ----
    // chunk c covers j = 32c..32c+31; lane holds j = 32c+off..+7 (elems 0-7)
    // and j = 32c+16+off..+7 (elems 8-15), matching 16-bit B 32x16 layout.
    v16h xreg[16];
    {
        const float* xr = x + (size_t)b_row * NX;
        #pragma unroll
        for (int c = 0; c < 16; ++c) {
            v4f f0 = *(const v4f*)(xr + 32 * c + off);
            v4f f1 = *(const v4f*)(xr + 32 * c + off + 4);
            v4f f2 = *(const v4f*)(xr + 32 * c + 16 + off);
            v4f f3 = *(const v4f*)(xr + 32 * c + 16 + off + 4);
            #pragma unroll
            for (int e = 0; e < 4; ++e) {
                xreg[c][e]      = (_Float16)f0[e];
                xreg[c][4 + e]  = (_Float16)f1[e];
                xreg[c][8 + e]  = (_Float16)f2[e];
                xreg[c][12 + e] = (_Float16)f3[e];
            }
        }
    }

    // ---- accumulator starts at bias[k] (D layout: VGPR g -> k = g + 8*half,
    //      lane column -> batch b_row) ----
    v8f acc;
    {
        v4f b0 = *(const v4f*)(bias + k0 + 8 * half);
        v4f b1 = *(const v4f*)(bias + k0 + 8 * half + 4);
        #pragma unroll
        for (int e = 0; e < 4; ++e) { acc[e] = b0[e]; acc[4 + e] = b1[e]; }
    }

    // staging decomposition for this thread
    const int sk = threadIdx.x & 3;
    const int sj = threadIdx.x >> 2;            // 0..63

    // prime the pipeline: slice 0 into buffer 0
    stage_slice(Q, 0, k0, sk, sj, ldsQ[0]);
    {   // prefetch slice 1 toward L2
        const float* qn = Q + (size_t)1 * (NX * NY) + k0;
        #pragma unroll
        for (int r = 0; r < 8; ++r)
            __builtin_prefetch(qn + (size_t)(sj + 64 * r) * NY + 4 * sk, 0, 0);
    }

    for (int i = 0; i < NX; ++i) {
        const float scale = x[(size_t)b_row * NX + i];

        __syncthreads();    // buffer i&1 fully staged; prev reads of (i+1)&1 done

        // ---- overlap: stage slice i+1 into the other buffer while the
        //      WMMA burst below consumes buffer i&1 ----
        if (i + 1 < NX) {
            stage_slice(Q, i + 1, k0, sk, sj, ldsQ[(i + 1) & 1]);
            if (i + 2 < NX) {
                const float* qn = Q + (size_t)(i + 2) * (NX * NY) + k0;
                #pragma unroll
                for (int r = 0; r < 8; ++r)
                    __builtin_prefetch(qn + (size_t)(sj + 64 * r) * NY + 4 * sk, 0, 0);
            }
        }

        // ---- T_i(tile) = Q_i^T x X^T : 16 x f16 WMMAs over j ----
        const _Float16* buf = ldsQ[i & 1];
        v8f t = {};
        #pragma unroll
        for (int c = 0; c < 16; ++c) {
            union { v16h v; v8h h[2]; } a;
            a.h[0] = *(const v8h*)(&buf[krow * LDS_PITCH + 32 * c + off]);
            a.h[1] = *(const v8h*)(&buf[krow * LDS_PITCH + 32 * c + 16 + off]);
            t = __builtin_amdgcn_wmma_f32_16x16x32_f16(
                    false, a.v, false, xreg[c], (short)0, t, false, false);
        }

        // ---- fold: acc += x[b,i] * (T_i + W[i,k])  (quadratic + linear) ----
        const float* wrow = W + (size_t)i * NY + k0 + 8 * half;
        v4f w0 = *(const v4f*)(wrow);
        v4f w1 = *(const v4f*)(wrow + 4);
        #pragma unroll
        for (int e = 0; e < 4; ++e) {
            acc[e]     = fmaf(scale, t[e]     + w0[e], acc[e]);
            acc[4 + e] = fmaf(scale, t[4 + e] + w1[e], acc[4 + e]);
        }
    }

    // ---- epilogue: write out[b_row, k0 + 8*half + 0..7] ----
    float* orow = out + (size_t)b_row * NY + k0 + 8 * half;
    v4f o0, o1;
    #pragma unroll
    for (int e = 0; e < 4; ++e) { o0[e] = acc[e]; o1[e] = acc[4 + e]; }
    *(v4f*)(orow)     = o0;
    *(v4f*)(orow + 4) = o1;
}

extern "C" void kernel_launch(void* const* d_in, const int* in_sizes, int n_in,
                              void* d_out, int out_size, void* d_ws, size_t ws_size,
                              hipStream_t stream) {
    (void)in_sizes; (void)n_in; (void)out_size; (void)d_ws; (void)ws_size;
    const float* x = (const float*)d_in[0];
    const float* W = (const float*)d_in[1];
    const float* Q = (const float*)d_in[2];
    const float* b = (const float*)d_in[3];
    float* out = (float*)d_out;

    dim3 grid(NBATCH / 128, NY / 16);   // (8, 32) = 256 blocks
    quad_fused_kernel<<<grid, 256, 0, stream>>>(x, W, Q, b, out);
}
